// MultiHeadLatentAttention_18425409700099
// MI455X (gfx1250) — compile-verified, matrix operand order checked by reading, not run on hardware
//
#include <hip/hip_runtime.h>
#include <math.h>

typedef unsigned short u16;
typedef __attribute__((ext_vector_type(16))) __bf16 bf16x16;
typedef __attribute__((ext_vector_type(8)))  float  v8f;
typedef __attribute__((ext_vector_type(4)))  int    v4i;

typedef __attribute__((address_space(1))) v4i g_v4i;   // global
typedef __attribute__((address_space(3))) v4i l_v4i;   // LDS

#if __has_builtin(__builtin_amdgcn_global_load_async_to_lds_b128)
#define HAVE_ASYNC_LDS 1
#endif

union Frag { bf16x16 v; uint4 q[2]; };

__device__ __forceinline__ u16 f32_bf16(float f) {
  union { float f; unsigned int u; } c; c.f = f;
  unsigned int u = c.u + 0x7FFFu + ((c.u >> 16) & 1u);   // round-to-nearest-even
  return (u16)(u >> 16);
}

// 16-byte global -> LDS copy. Uses the CDNA5 async global->LDS path when the
// toolchain exposes it (tracked by ASYNCcnt), else VGPR bounce.
__device__ __forceinline__ void copy16_g2l(const u16* g, u16* l) {
#ifdef HAVE_ASYNC_LDS
  __builtin_amdgcn_global_load_async_to_lds_b128(
      (g_v4i*)(unsigned long long)(const void*)g,
      (l_v4i*)(unsigned)(unsigned long long)(void*)l, 0, 0);
#else
  *(uint4*)l = *(const uint4*)g;
#endif
}

__device__ __forceinline__ void wait_async_lds() {
#ifdef HAVE_ASYNC_LDS
#if __has_builtin(__builtin_amdgcn_s_wait_asynccnt)
  __builtin_amdgcn_s_wait_asynccnt(0);
#else
  asm volatile("s_wait_asynccnt 0x0" ::: "memory");
#endif
#endif
}

// Load one 16x32 bf16 A/B operand fragment from LDS.
// Lanes 0-15: row r, K = {b..b+7, b+16..b+23}; lanes 16-31: K = +8; b = half*8.
__device__ __forceinline__ Frag ld_frag_lds(const u16* p) {
  Frag f;
  f.q[0] = *(const uint4*)(p);
  f.q[1] = *(const uint4*)(p + 16);
  return f;
}

__global__ void cast_f32_bf16_kernel(const float* __restrict__ src,
                                     u16* __restrict__ dst, int n) {
  int i = blockIdx.x * blockDim.x + threadIdx.x;
  if (i < n) dst[i] = f32_bf16(src[i]);
}

// src f32 [K][N] -> dst bf16 [N][K] (weights pre-transposed once so GEMM
// B-tiles become straight row-major copies; K,N multiples of 32 here).
__global__ __launch_bounds__(256) void cast_transpose_kernel(
    const float* __restrict__ src, u16* __restrict__ dst, int K, int N) {
  __shared__ float tile[32][33];
  int kb = blockIdx.y << 5, nb = blockIdx.x << 5;
  int tx = threadIdx.x & 31, ty = threadIdx.x >> 5;
  #pragma unroll
  for (int i = ty; i < 32; i += 8) tile[i][tx] = src[(size_t)(kb + i) * N + nb + tx];
  __syncthreads();
  #pragma unroll
  for (int i = ty; i < 32; i += 8)
    dst[(size_t)(nb + i) * K + kb + tx] = f32_bf16(tile[tx][i]);
}

// bf16 [R][C] -> bf16 [C][R] (for V; R,C multiples of 32).
__global__ __launch_bounds__(256) void transpose_bf16_kernel(
    const u16* __restrict__ src, u16* __restrict__ dst, int R, int C) {
  __shared__ u16 tile[32][40];
  int rb = blockIdx.y << 5, cb = blockIdx.x << 5;
  int tx = threadIdx.x & 31, ty = threadIdx.x >> 5;
  #pragma unroll
  for (int i = ty; i < 32; i += 8) tile[i][tx] = src[(size_t)(rb + i) * C + cb + tx];
  __syncthreads();
  #pragma unroll
  for (int i = ty; i < 32; i += 8)
    dst[(size_t)(cb + i) * R + rb + tx] = tile[tx][i];
}

// ---------------------------------------------------------------------------
// C[M,N] = alpha * (A[M,K] @ Bt[N,K]^T + bias[N]); A,Bt bf16 row-major.
// 256 threads = 8 waves; block tile 128x128, K-step 32, double-buffered LDS,
// async global->LDS staging. Wave owns 32x64 = 2x4 WMMA tiles.
// ---------------------------------------------------------------------------
template <bool OUT_F32>
__global__ __launch_bounds__(256) void gemm_bf16_kernel(
    const u16* __restrict__ A, const u16* __restrict__ Bt,
    const float* __restrict__ bias, void* __restrict__ C,
    int M, int N, int K, float alpha)
{
  __shared__ __align__(16) u16 sA[2 * 128 * 40];  // [m][k], k padded 32->40
  __shared__ __align__(16) u16 sB[2 * 128 * 40];  // [n][k]

  const int tid  = threadIdx.x;
  const int lane = tid & 31;
  const int wv   = tid >> 5;
  const int wm   = (wv & 3) << 5;
  const int wn   = (wv >> 2) << 6;
  const int r    = lane & 15;
  const int half = lane >> 4;
  const int m0   = blockIdx.y << 7;
  const int n0   = blockIdx.x << 7;

  auto stage = [&](int k0, int b) {
    u16* dA = sA + b * (128 * 40);
    u16* dB = sB + b * (128 * 40);
    #pragma unroll
    for (int c = tid; c < 512; c += 256) {
      int row = c >> 2, ko = (c & 3) << 3;
      copy16_g2l(A + (size_t)(m0 + row) * K + k0 + ko, dA + row * 40 + ko);
    }
    if (n0 + 128 <= N) {
      #pragma unroll
      for (int c = tid; c < 512; c += 256) {
        int row = c >> 2, ko = (c & 3) << 3;
        copy16_g2l(Bt + (size_t)(n0 + row) * K + k0 + ko, dB + row * 40 + ko);
      }
    } else {  // N tail (only the N=64 GEMM): zero-fill out-of-range rows
      #pragma unroll
      for (int c = tid; c < 512; c += 256) {
        int row = c >> 2, ko = (c & 3) << 3;
        uint4 t = {0u, 0u, 0u, 0u};
        if (n0 + row < N)
          t = *(const uint4*)(Bt + (size_t)(n0 + row) * K + k0 + ko);
        *(uint4*)(dB + row * 40 + ko) = t;
      }
    }
  };

  const v8f vzero = {};
  v8f acc[2][4];
  #pragma unroll
  for (int mi = 0; mi < 2; ++mi)
    #pragma unroll
    for (int ni = 0; ni < 4; ++ni) acc[mi][ni] = vzero;

  const int nstage = K >> 5;
  stage(0, 0);
  wait_async_lds();
  __syncthreads();

  for (int s = 0; s < nstage; ++s) {
    const int buf = s & 1;
    if (s + 1 < nstage) stage((s + 1) << 5, buf ^ 1);  // prefetch next stage

    const u16* bA = sA + buf * (128 * 40);
    const u16* bB = sB + buf * (128 * 40);
    Frag a[2], b[4];
    #pragma unroll
    for (int mi = 0; mi < 2; ++mi)
      a[mi] = ld_frag_lds(bA + (wm + mi * 16 + r) * 40 + half * 8);
    #pragma unroll
    for (int ni = 0; ni < 4; ++ni)
      b[ni] = ld_frag_lds(bB + (wn + ni * 16 + r) * 40 + half * 8);
    #pragma unroll
    for (int mi = 0; mi < 2; ++mi)
      #pragma unroll
      for (int ni = 0; ni < 4; ++ni)
        acc[mi][ni] = __builtin_amdgcn_wmma_f32_16x16x32_bf16(
            false, a[mi].v, false, b[ni].v, (short)0, acc[mi][ni],
            false, false);

    wait_async_lds();
    __syncthreads();
  }

  // C layout: VGPR v, lanes 0-15 -> (M=v, N=lane); lanes 16-31 -> (M=v+8).
  #pragma unroll
  for (int mi = 0; mi < 2; ++mi)
    #pragma unroll
    for (int ni = 0; ni < 4; ++ni) {
      int col = n0 + wn + ni * 16 + r;
      if (col >= N) continue;
      float bv = bias[col];
      #pragma unroll
      for (int v = 0; v < 8; ++v) {
        int row = m0 + wm + mi * 16 + v + (half << 3);
        float val = alpha * (acc[mi][ni][v] + bv);
        if (OUT_F32) ((float*)C)[(size_t)row * N + col] = val;
        else         ((u16*)C)[(size_t)row * N + col]   = f32_bf16(val);
      }
    }
}

// ---------------------------------------------------------------------------
// Flash attention over [content(128) | rotary(64)] dims, softmax over all
// 2048 keys (non-causal). 256 threads = 8 waves; q-tile 128 (16 rows/wave),
// key tile 64, double-buffered async K/V staging. V pre-transposed [d][s].
// ---------------------------------------------------------------------------
__global__ __launch_bounds__(256) void mla_flash_kernel(
    const u16* __restrict__ Q,  const u16* __restrict__ Qr,
    const u16* __restrict__ Kc, const u16* __restrict__ Kr,
    const u16* __restrict__ Vt, u16* __restrict__ O)
{
  extern __shared__ char smem_raw[];
  u16* sQ  = (u16*)smem_raw;       // 128 x 200
  u16* sK  = sQ  + 128 * 200;      // 2 x (64 x 200)
  u16* sVt = sK  + 2 * 64 * 200;   // 2 x (128 x 72)   [d][key]
  u16* sP  = sVt + 2 * 128 * 72;   // 128 x 72 (wave-private rows)

  const int tid  = threadIdx.x;
  const int lane = tid & 31;
  const int wv   = tid >> 5;
  const int r    = lane & 15;
  const int half = lane >> 4;
  const int h    = blockIdx.y;
  const int q0   = blockIdx.x << 7;
  const int rb   = wv << 4;

  auto stageKV = [&](int k0, int b) {
    u16* dK = sK + b * (64 * 200);
    u16* dV = sVt + b * (128 * 72);
    #pragma unroll 2
    for (int c = tid; c < 64 * 24; c += 256) {
      int row = c / 24, j = c % 24;
      const u16* src = (j < 16)
          ? Kc + (size_t)(k0 + row) * 4096 + h * 128 + j * 8
          : Kr + (size_t)(k0 + row) * 64 + (j - 16) * 8;
      copy16_g2l(src, dK + row * 200 + j * 8);
    }
    #pragma unroll
    for (int c = tid; c < 128 * 8; c += 256) {
      int d = c >> 3, ko = (c & 7) << 3;
      copy16_g2l(Vt + (size_t)((h << 7) + d) * 2048 + k0 + ko, dV + d * 72 + ko);
    }
  };

  // Stage Qc (cols 0..127 content, 128..191 rotary) + first K/V tile.
  #pragma unroll 2
  for (int c = tid; c < 128 * 24; c += 256) {
    int row = c / 24, j = c % 24;
    const u16* src = (j < 16)
        ? Q  + (size_t)(q0 + row) * 4096 + h * 128 + j * 8
        : Qr + (size_t)(q0 + row) * 2048 + h * 64 + (j - 16) * 8;
    copy16_g2l(src, sQ + row * 200 + j * 8);
  }
  stageKV(0, 0);

  const v8f vzero = {};
  v8f o[8];
  #pragma unroll
  for (int d = 0; d < 8; ++d) o[d] = vzero;
  float m_i[8], l_i[8];
  #pragma unroll
  for (int v = 0; v < 8; ++v) { m_i[v] = -__builtin_inff(); l_i[v] = 0.f; }

  wait_async_lds();
  __syncthreads();

  for (int k0 = 0; k0 < 2048; k0 += 64) {
    const int buf = (k0 >> 6) & 1;
    if (k0 + 64 < 2048) stageKV(k0 + 64, buf ^ 1);  // prefetch next tile
    const u16* bK = sK + buf * (64 * 200);
    const u16* bV = sVt + buf * (128 * 72);

    // S = Qc @ Kc^T : 16 q-rows x 64 keys per wave, K-dim 192 = 6 steps.
    v8f s[4] = {vzero, vzero, vzero, vzero};
    #pragma unroll
    for (int kk = 0; kk < 6; ++kk) {
      Frag a = ld_frag_lds(sQ + (rb + r) * 200 + kk * 32 + half * 8);
      #pragma unroll
      for (int nt = 0; nt < 4; ++nt) {
        Frag b = ld_frag_lds(bK + (nt * 16 + r) * 200 + kk * 32 + half * 8);
        s[nt] = __builtin_amdgcn_wmma_f32_16x16x32_bf16(
            false, a.v, false, b.v, (short)0, s[nt], false, false);
      }
    }

    // Online softmax: acc elem v on lane (r,half) is row rb+v+8*half,
    // col nt*16+r; row reductions = shfl within each 16-lane half.
    #pragma unroll
    for (int v = 0; v < 8; ++v) {
      float mx = fmaxf(fmaxf(s[0][v], s[1][v]), fmaxf(s[2][v], s[3][v]));
      #pragma unroll
      for (int off = 1; off < 16; off <<= 1)
        mx = fmaxf(mx, __shfl_xor(mx, off, 32));
      float mn = fmaxf(m_i[v], mx);
      float sc = __expf(m_i[v] - mn);
      float rs = 0.f;
      u16* prow = sP + (rb + v + (half << 3)) * 72;
      #pragma unroll
      for (int nt = 0; nt < 4; ++nt) {
        float p = __expf(s[nt][v] - mn);
        rs += p;
        prow[nt * 16 + r] = f32_bf16(p);
      }
      #pragma unroll
      for (int off = 1; off < 16; off <<= 1)
        rs += __shfl_xor(rs, off, 32);
      l_i[v] = l_i[v] * sc + rs;
      m_i[v] = mn;
      #pragma unroll
      for (int d = 0; d < 8; ++d) o[d][v] *= sc;
    }

    // O += P @ V : A = P(16x64), B = Vt tile [d][key]; in-wave DS ordering
    // covers sP write->read (rows are wave-private).
    #pragma unroll
    for (int kk = 0; kk < 2; ++kk) {
      Frag p = ld_frag_lds(sP + (rb + r) * 72 + kk * 32 + half * 8);
      #pragma unroll
      for (int d = 0; d < 8; ++d) {
        Frag vb = ld_frag_lds(bV + (d * 16 + r) * 72 + kk * 32 + half * 8);
        o[d] = __builtin_amdgcn_wmma_f32_16x16x32_bf16(
            false, p.v, false, vb.v, (short)0, o[d], false, false);
      }
    }

    wait_async_lds();
    __syncthreads();
  }

  #pragma unroll
  for (int d = 0; d < 8; ++d) {
    int col = (h << 7) + d * 16 + r;
    #pragma unroll
    for (int v = 0; v < 8; ++v) {
      int row = q0 + rb + v + (half << 3);
      O[(size_t)row * 4096 + col] = f32_bf16(o[d][v] / l_i[v]);
    }
  }
}

// ---------------------------------------------------------------------------
extern "C" void kernel_launch(void* const* d_in, const int* in_sizes, int n_in,
                              void* d_out, int out_size, void* d_ws, size_t ws_size,
                              hipStream_t stream) {
  const float* seq   = (const float*)d_in[0];
  const float* w_dq  = (const float*)d_in[1];
  const float* b_dq  = (const float*)d_in[2];
  const float* w_uq  = (const float*)d_in[3];
  const float* b_uq  = (const float*)d_in[4];
  const float* w_dkv = (const float*)d_in[5];
  const float* b_dkv = (const float*)d_in[6];
  const float* w_uk  = (const float*)d_in[7];
  const float* b_uk  = (const float*)d_in[8];
  const float* w_uv  = (const float*)d_in[9];
  const float* b_uv  = (const float*)d_in[10];
  const float* w_rq  = (const float*)d_in[11];
  const float* b_rq  = (const float*)d_in[12];
  const float* w_rk  = (const float*)d_in[13];
  const float* b_rk  = (const float*)d_in[14];
  const float* w_out = (const float*)d_in[15];
  const float* b_out = (const float*)d_in[16];

  const int S = 2048, DE = 4096, DC = 512, DC1 = 1536, DM = 4096, HDR = 2048, DR = 64;
  const float scaler = 1.0f / sqrtf(192.0f);   // 1/sqrt(D_HEAD + D_R)

  size_t off = 0;
  auto carve = [&](size_t elems) -> u16* {
    u16* p = (u16*)((char*)d_ws + off);
    off += ((elems * sizeof(u16) + 255) & ~(size_t)255);
    return p;
  };
  u16* seq_bf  = carve((size_t)S * DE);
  u16* wdq_t   = carve((size_t)DE * DC1);   // transposed weights: [N][K]
  u16* wuq_t   = carve((size_t)DC1 * DM);
  u16* wdkv_t  = carve((size_t)DE * DC);
  u16* wuk_t   = carve((size_t)DC * DM);
  u16* wuv_t   = carve((size_t)DC * DM);
  u16* wrq_t   = carve((size_t)DC1 * HDR);
  u16* wrk_t   = carve((size_t)DE * DR);
  u16* wout_t  = carve((size_t)DM * DM);
  u16* cq_bf   = carve((size_t)S * DC1);
  u16* ckv_bf  = carve((size_t)S * DC);
  u16* q_bf    = carve((size_t)S * DM);
  u16* qr_bf   = carve((size_t)S * HDR);
  u16* k_bf    = carve((size_t)S * DM);
  u16* v_bf    = carve((size_t)S * DM);
  u16* vt_bf   = carve((size_t)DM * S);     // V transposed: [h*128+d][s]
  u16* kr_bf   = carve((size_t)S * DR);
  u16* ao_bf   = carve((size_t)S * DM);

  cast_f32_bf16_kernel<<<dim3((unsigned)((S * DE + 255) / 256)), dim3(256), 0,
                         stream>>>(seq, seq_bf, S * DE);
  auto castT = [&](const float* src, u16* dst, int K, int N) {
    cast_transpose_kernel<<<dim3(N / 32, K / 32), dim3(256), 0, stream>>>(
        src, dst, K, N);
  };
  castT(w_dq,  wdq_t,  DE,  DC1);
  castT(w_uq,  wuq_t,  DC1, DM);
  castT(w_dkv, wdkv_t, DE,  DC);
  castT(w_uk,  wuk_t,  DC,  DM);
  castT(w_uv,  wuv_t,  DC,  DM);
  castT(w_rq,  wrq_t,  DC1, HDR);
  castT(w_rk,  wrk_t,  DE,  DR);
  castT(w_out, wout_t, DM,  DM);

  auto gemm = [&](const u16* A, const u16* Bt, const float* bias, void* C,
                  int M, int N, int K, float alpha, bool outF32) {
    dim3 g((N + 127) / 128, (M + 127) / 128);
    if (outF32)
      gemm_bf16_kernel<true><<<g, 256, 0, stream>>>(A, Bt, bias, C, M, N, K, alpha);
    else
      gemm_bf16_kernel<false><<<g, 256, 0, stream>>>(A, Bt, bias, C, M, N, K, alpha);
  };

  // Low-rank query path (query scale folded into Q / Qr).
  gemm(seq_bf, wdq_t,  b_dq,  cq_bf,  S, DC1, DE,  1.0f,   false);
  gemm(cq_bf,  wuq_t,  b_uq,  q_bf,   S, DM,  DC1, scaler, false);
  gemm(cq_bf,  wrq_t,  b_rq,  qr_bf,  S, HDR, DC1, scaler, false);
  // Low-rank KV path.
  gemm(seq_bf, wdkv_t, b_dkv, ckv_bf, S, DC,  DE,  1.0f,   false);
  gemm(ckv_bf, wuk_t,  b_uk,  k_bf,   S, DM,  DC,  1.0f,   false);
  gemm(ckv_bf, wuv_t,  b_uv,  v_bf,   S, DM,  DC,  1.0f,   false);
  gemm(seq_bf, wrk_t,  b_rk,  kr_bf,  S, DR,  DE,  1.0f,   false);
  // V -> V^T so attention's PV operand stages as straight copies.
  transpose_bf16_kernel<<<dim3(DM / 32, S / 32), dim3(256), 0, stream>>>(
      v_bf, vt_bf, S, DM);

  // Flash attention: grid = (q-tiles, heads), ~154 KB dynamic LDS.
  size_t shmem = (size_t)(128 * 200 + 2 * 64 * 200 + 2 * 128 * 72 + 128 * 72)
                 * sizeof(u16);
  mla_flash_kernel<<<dim3(16, 32), dim3(256), shmem, stream>>>(
      q_bf, qr_bf, k_bf, kr_bf, vt_bf, ao_bf);

  // Output projection, fp32 straight to d_out.
  gemm(ao_bf, wout_t, b_out, d_out, S, DM, DM, 1.0f, true);
}
